// QLinearLayer_30820685316556
// MI455X (gfx1250) — matrix operand order repaired and checked by reading, not compiled, not run on hardware
//
#include <hip/hip_runtime.h>

typedef __attribute__((ext_vector_type(16))) _Float16 v16h;
typedef __attribute__((ext_vector_type(8)))  _Float16 v8h;
typedef __attribute__((ext_vector_type(8)))  float    v8f;
typedef __attribute__((ext_vector_type(4)))  unsigned int v4u;
typedef __attribute__((ext_vector_type(8)))  int v8i;
typedef __attribute__((ext_vector_type(4)))  int v4i;

union Frag { v16h v; v8h h[2]; };

#define BM 128
#define BN 128
#define BK 32
#define LDST 40   // LDS row stride in f16 (80B): TDM pad 4 DW per 16 DW row

// ---------------- NVFP4 numerics helpers ----------------

__device__ inline float e4m3_round(float x) {
    if (!(x > 0.f)) return 0.f;
    if (x >= 448.f) return 448.f;
    unsigned u = __float_as_uint(x);
    const unsigned keep = 20;              // 23 - 3 mantissa bits
    unsigned lsb = (u >> keep) & 1u;
    u += 0x7FFFFu + lsb;                   // round to nearest even
    u &= ~((1u << keep) - 1u);
    float r = __uint_as_float(u);
    if (r < 0.015625f) {                   // below 2^-6: subnormal grid = 2^-9
        r = rintf(r * 512.f) * (1.f / 512.f);
    }
    return fminf(r, 448.f);
}

__device__ inline float fp4_round(float v) {
    float a = fabsf(v);
    float s = (v < 0.f) ? -1.f : 1.f;
    float q;
    if      (a < 0.25f) q = 0.0f;
    else if (a < 0.75f) q = 0.5f;
    else if (a < 1.25f) q = 1.0f;
    else if (a < 1.75f) q = 1.5f;
    else if (a < 2.5f)  q = 2.0f;
    else if (a < 3.5f)  q = 3.0f;
    else if (a < 5.0f)  q = 4.0f;
    else                q = 6.0f;
    return s * q;
}

__device__ inline float tensor_scale_from_amax(unsigned amax_bits) {
    float ts = __uint_as_float(amax_bits) * (1.0f / (6.0f * 448.0f));
    return (ts > 0.f) ? ts : 1.f;
}

// ---------------- prepass kernels ----------------

__global__ void init_amax_kernel(unsigned* amax) {
    amax[0] = 0u;
    amax[1] = 0u;
}

__global__ void amax_kernel(const __bf16* __restrict__ p, size_t n,
                            unsigned* __restrict__ out) {
    __shared__ float red[256];
    float m = 0.f;
    for (size_t i = (size_t)blockIdx.x * blockDim.x + threadIdx.x; i < n;
         i += (size_t)gridDim.x * blockDim.x) {
        m = fmaxf(m, fabsf((float)p[i]));
    }
    red[threadIdx.x] = m;
    __syncthreads();
    for (int s = 128; s > 0; s >>= 1) {
        if ((int)threadIdx.x < s)
            red[threadIdx.x] = fmaxf(red[threadIdx.x], red[threadIdx.x + s]);
        __syncthreads();
    }
    if (threadIdx.x == 0) atomicMax(out, __float_as_uint(red[0]));
}

// One thread per 16-element NVFP4 block. Output is q*sf in f16 — exact
// (q needs <=1 mantissa bit, sf is E4M3 with 3 -> product <=5 bits, range
// [2^-10, 2688] inside f16). The per-tensor fp32 scale ts is factored out
// and applied in the GEMM epilogue.
__global__ void qdq_kernel(const __bf16* __restrict__ in, _Float16* __restrict__ out,
                           const unsigned* __restrict__ amaxbuf, int which,
                           size_t nblk) {
    size_t b = (size_t)blockIdx.x * blockDim.x + threadIdx.x;
    if (b >= nblk) return;
    float ts = tensor_scale_from_amax(amaxbuf[which]);

    const __bf16* src = in + b * 16;
    float v[16];
    float bmax = 0.f;
#pragma unroll
    for (int i = 0; i < 16; i++) {
        v[i] = (float)src[i];
        bmax = fmaxf(bmax, fabsf(v[i]));
    }
    float sf  = e4m3_round((bmax * (1.f / 6.f)) / ts);
    float sft = sf * ts;
    float inv = (sft > 0.f) ? (1.f / sft) : 0.f;

    _Float16* dst = out + b * 16;
#pragma unroll
    for (int i = 0; i < 16; i++) {
        dst[i] = (_Float16)(fp4_round(v[i] * inv) * sf);
    }
}

// ---------------- TDM tile load (Tensor Data Mover) ----------------
// 2D tile: tile_dim0 = BK (32 f16 = 16 DWORDs per row), tile_dim1 = 128 rows,
// tensor_dim0_stride = K. LDS padding 4 DWORDs per 16 DWORDs -> 80B row stride
// in LDS == LDST(40) f16, matching the WMMA fragment indexing below.
__device__ __forceinline__ void tdm_load_tile(unsigned lds_byte_addr,
                                              const _Float16* tile_gaddr,
                                              unsigned tensor_rows,
                                              unsigned tensor_k) {
    unsigned long long ga = (unsigned long long)(uintptr_t)tile_gaddr;
    // Group 0: count=1 | lds_addr | global_addr[56:0] | type=2
    v4u g0 = { 1u,
               lds_byte_addr,
               (unsigned)ga,
               ((unsigned)(ga >> 32) & 0x01FFFFFFu) | (2u << 30) };
    // Group 1 (256b): flags | tensor_dim0/1 | tile_dim0/1/2 | dim0_stride
    // flags: data_size=1(2B)<<16 | pad_enable<<20 | pad_interval=3(16DW)<<22
    //        | pad_amount=3(4DW)<<25  == 0x06D10000
    union { unsigned long long q[4]; v8i v; } g1;
    g1.q[0] = 0x06D10000ull
            | ((unsigned long long)(tensor_k & 0xFFFFu) << 48);       // dim0 lo
    g1.q[1] = (unsigned long long)(tensor_k >> 16)                    // dim0 hi
            | ((unsigned long long)tensor_rows << 16)                 // dim1
            | ((unsigned long long)BK << 48);                         // tile_dim0
    g1.q[2] = (unsigned long long)BM                                  // tile_dim1
            | ((unsigned long long)tensor_k << 32);                   // dim0_stride lo
    g1.q[3] = 0ull;                                                   // stride hi, dim1_stride
    v4i zero4 = { 0, 0, 0, 0 };
    v8i zero8 = { 0, 0, 0, 0, 0, 0, 0, 0 };
    __builtin_amdgcn_tensor_load_to_lds(g0, g1.v, zero4, zero4, zero8, 0);
}

// ---------------- WMMA GEMM: C[M,N] = (tsx*tsw) * A[M,K] * B[N,K]^T --------

__global__ __launch_bounds__(256)
void gemm_f16_wmma_kernel(const _Float16* __restrict__ A,   // [M,K] q*sf (x)
                          const _Float16* __restrict__ B,   // [N,K] q*sf (w)
                          __bf16* __restrict__ C,           // [M,N]
                          const unsigned* __restrict__ amaxbuf,
                          int M, int N, int K) {
    __shared__ __attribute__((aligned(16))) _Float16 sA[2][BM * LDST];
    __shared__ __attribute__((aligned(16))) _Float16 sB[2][BN * LDST];

    const int t    = threadIdx.x;
    const int lane = t & 31;
    const int wave = t >> 5;          // 8 waves
    const int wm   = wave & 1;        // 2 wave-rows  -> 64 M-rows each
    const int wn   = wave >> 1;       // 4 wave-cols  -> 32 N-cols each
    const int lrow = lane & 15;
    const int lhi  = lane >> 4;       // 0 or 1

    const size_t bm = (size_t)blockIdx.x * BM;
    const size_t bn = (size_t)blockIdx.y * BN;

    // Deferred per-tensor scales (exact NVFP4 factoring).
    const float outscale = tensor_scale_from_amax(amaxbuf[0]) *
                           tensor_scale_from_amax(amaxbuf[1]);

    const unsigned ldsA0 = (unsigned)(uintptr_t)&sA[0][0];
    const unsigned ldsA1 = (unsigned)(uintptr_t)&sA[1][0];
    const unsigned ldsB0 = (unsigned)(uintptr_t)&sB[0][0];
    const unsigned ldsB1 = (unsigned)(uintptr_t)&sB[1][0];

    v8f acc[4][2];
#pragma unroll
    for (int i = 0; i < 4; i++)
#pragma unroll
        for (int j = 0; j < 2; j++) acc[i][j] = (v8f)0.f;

    const int nIter = K / BK;

    // Prologue: DMA the first K-tile into buffer 0 (one wave issues per WG).
    if (wave == 0) {
        tdm_load_tile(ldsA0, A + bm * (size_t)K, (unsigned)M, (unsigned)K);
        tdm_load_tile(ldsB0, B + bn * (size_t)K, (unsigned)N, (unsigned)K);
    }

    for (int it = 0; it < nIter; ++it) {
        const int cur = it & 1;
        if (wave == 0) {
            if (it + 1 < nIter) {
                const int k1 = (it + 1) * BK;
                tdm_load_tile(cur ? ldsA0 : ldsA1,
                              A + bm * (size_t)K + k1, (unsigned)M, (unsigned)K);
                tdm_load_tile(cur ? ldsB0 : ldsB1,
                              B + bn * (size_t)K + k1, (unsigned)N, (unsigned)K);
                // 4 outstanding; oldest 2 (current buffer) must be complete.
                __builtin_amdgcn_s_wait_tensorcnt(2);
            } else {
                __builtin_amdgcn_s_wait_tensorcnt(0);
            }
        }
        __syncthreads();   // current buffer visible to all waves

        const _Float16* pa = &sA[cur][0];
        const _Float16* pb = &sB[cur][0];

        // B fragments: column lrow, K = 16*lhi .. +16 contiguous.
        Frag bf0, bf1;
        {
            int col0 = wn * 32 + lrow;
            int col1 = wn * 32 + 16 + lrow;
            bf0.h[0] = *(const v8h*)&pb[col0 * LDST + lhi * 16];
            bf0.h[1] = *(const v8h*)&pb[col0 * LDST + lhi * 16 + 8];
            bf1.h[0] = *(const v8h*)&pb[col1 * LDST + lhi * 16];
            bf1.h[1] = *(const v8h*)&pb[col1 * LDST + lhi * 16 + 8];
        }
        // A fragments: row lrow, K = 16*lhi .. +16 contiguous.
#pragma unroll
        for (int i = 0; i < 4; i++) {
            Frag af;
            int row = wm * 64 + i * 16 + lrow;
            af.h[0] = *(const v8h*)&pa[row * LDST + lhi * 16];
            af.h[1] = *(const v8h*)&pa[row * LDST + lhi * 16 + 8];
            acc[i][0] = __builtin_amdgcn_wmma_f32_16x16x32_f16(
                false, af.v, false, bf0.v, (short)0, acc[i][0], false, false);
            acc[i][1] = __builtin_amdgcn_wmma_f32_16x16x32_f16(
                false, af.v, false, bf1.v, (short)0, acc[i][1], false, false);
        }
        __syncthreads();   // all reads done before this buffer is re-DMA'd
    }

    // Epilogue: apply per-tensor scale product; C/D f32 layout:
    // VGPR r holds M = r + 8*(lane/16), N = lane%16.
#pragma unroll
    for (int i = 0; i < 4; i++) {
#pragma unroll
        for (int j = 0; j < 2; j++) {
#pragma unroll
            for (int r = 0; r < 8; r++) {
                size_t row = bm + wm * 64 + i * 16 + r + lhi * 8;
                size_t col = bn + wn * 32 + j * 16 + lrow;
                C[row * (size_t)N + col] = (__bf16)(acc[i][j][r] * outscale);
            }
        }
    }
}

// ---------------- launch ----------------

extern "C" void kernel_launch(void* const* d_in, const int* in_sizes, int n_in,
                              void* d_out, int out_size, void* d_ws, size_t ws_size,
                              hipStream_t stream) {
    const int K = 4096;
    const __bf16* x = (const __bf16*)d_in[0];
    const __bf16* w = (const __bf16*)d_in[1];
    const int M = in_sizes[0] / K;     // 16384
    const int N = in_sizes[1] / K;     // 4096

    char* ws = (char*)d_ws;
    unsigned* amax = (unsigned*)ws;                                  // 2 x u32
    _Float16* dqx = (_Float16*)(ws + 256);                           // M*K f16
    _Float16* dqw = (_Float16*)(ws + 256 + (size_t)M * K * 2);       // N*K f16

    init_amax_kernel<<<1, 1, 0, stream>>>(amax);
    amax_kernel<<<2048, 256, 0, stream>>>(x, (size_t)M * K, amax + 0);
    amax_kernel<<<2048, 256, 0, stream>>>(w, (size_t)N * K, amax + 1);

    size_t nbx = (size_t)M * K / 16;
    size_t nbw = (size_t)N * K / 16;
    qdq_kernel<<<(int)((nbx + 255) / 256), 256, 0, stream>>>(x, dqx, amax, 0, nbx);
    qdq_kernel<<<(int)((nbw + 255) / 256), 256, 0, stream>>>(w, dqw, amax, 1, nbw);

    dim3 grid(M / BM, N / BN);
    gemm_f16_wmma_kernel<<<grid, 256, 0, stream>>>(dqx, dqw, (__bf16*)d_out,
                                                   amax, M, N, K);
}
